// TorchEinsumScatter_18150531793005
// MI455X (gfx1250) — matrix operand order repaired and checked by reading, not compiled, not run on hardware
//
#include <hip/hip_runtime.h>

// Problem constants (from the reference setup_inputs)
constexpr int kB = 8;     // batch
constexpr int kE = 8;     // experts
constexpr int kK = 512;   // tokens per (b,e)
constexpr int kJ = 512;   // inner (contraction) dim
constexpr int kI = 512;   // output feature dim
constexpr int kT = 2048;  // scatter segments

typedef _Float16 v8h  __attribute__((ext_vector_type(8)));
typedef _Float16 v16h __attribute__((ext_vector_type(16)));
typedef float    v8f  __attribute__((ext_vector_type(8)));

// Tile config: block = 256 threads = 8 waves (wave32).
// Block tile: 128 (k) x 64 (i). Wave tile: 32x32 -> 2x2 WMMA accumulators.
// J is processed in chunks of 64 (two K=32 WMMA steps per chunk).
constexpr int TM  = 128;       // k-rows per block
constexpr int TN  = 64;        // i-cols per block
constexpr int TJ  = 64;        // j-chunk
constexpr int SAJ = TJ + 8;    // padded LDS row stride (halves); 144B, 16B-aligned

__global__ __launch_bounds__(256)
void zero_out_kernel(float4* __restrict__ out, int n4) {
  int i = blockIdx.x * blockDim.x + threadIdx.x;
  if (i < n4) out[i] = make_float4(0.f, 0.f, 0.f, 0.f);
}

// Split fp32 into f16 hi + f16 residual (x ~= hi + lo, ~22-bit mantissa total).
static __device__ inline void split_f32(float v, _Float16& h, _Float16& l) {
  h = (_Float16)v;
  l = (_Float16)(v - (float)h);
}

// Assemble a 16-bit WMMA operand fragment from two contiguous 16-byte LDS runs.
static __device__ inline v16h frag16(const _Float16* p0, const _Float16* p1) {
  v8h x = *reinterpret_cast<const v8h*>(p0);
  v8h y = *reinterpret_cast<const v8h*>(p1);
  v16h r;
#pragma unroll
  for (int q = 0; q < 8; ++q) { r[q] = x[q]; r[q + 8] = y[q]; }
  return r;
}

__global__ __launch_bounds__(256)
void gemm_scatter_kernel(const float* __restrict__ Y,
                         const float* __restrict__ W,
                         const int*   __restrict__ Ind,
                         float* __restrict__ out) {
  // f16 hi/lo tiles. sA: [k][j] row-major; sB: [i][j] (W chunk transposed) so
  // both A and B fragments are contiguous ds_load_b128 pairs.
  __shared__ alignas(16) _Float16 sAhi[TM][SAJ];
  __shared__ alignas(16) _Float16 sAlo[TM][SAJ];
  __shared__ alignas(16) _Float16 sBhi[TN][SAJ];
  __shared__ alignas(16) _Float16 sBlo[TN][SAJ];
  __shared__ int sInd[TM];

  const int be = blockIdx.z;
  const int b  = be / kE;
  const int e  = be % kE;
  const int k0 = blockIdx.y * TM;
  const int i0 = blockIdx.x * TN;

  const int tid  = threadIdx.x;
  const int wave = tid >> 5;      // 0..7
  const int lane = tid & 31;
  const int wm   = wave & 3;      // wave's M (k) tile: wm*32
  const int wn   = wave >> 2;     // wave's N (i) tile: wn*32
  const int ln   = lane & 15;
  const int lh   = lane >> 4;     // lane half (selects K halves / M+8 in C)

  if (tid < TM) sInd[tid] = Ind[(be * kK) + k0 + tid];

  v8f acc00 = {}, acc01 = {}, acc10 = {}, acc11 = {};

  for (int jc = 0; jc < kJ / TJ; ++jc) {
    const int j0 = jc * TJ;

    // ---- Stage Y tile (128 x 64 f32) split into f16 hi/lo: 8 float4/thread.
#pragma unroll
    for (int v = 0; v < 8; ++v) {
      int fi  = v * 256 + tid;          // float4 index 0..2047
      int row = fi >> 4;                // 16 float4 per row
      int c4  = (fi & 15) << 2;
      float4 vy = *reinterpret_cast<const float4*>(
          &Y[((size_t)(be * kK + k0 + row)) * kJ + j0 + c4]);
      split_f32(vy.x, sAhi[row][c4 + 0], sAlo[row][c4 + 0]);
      split_f32(vy.y, sAhi[row][c4 + 1], sAlo[row][c4 + 1]);
      split_f32(vy.z, sAhi[row][c4 + 2], sAlo[row][c4 + 2]);
      split_f32(vy.w, sAhi[row][c4 + 3], sAlo[row][c4 + 3]);
    }

    // ---- Stage W tile (64 x 64 f32) transposed into sB[i][j]: 4 float4/thread.
#pragma unroll
    for (int v = 0; v < 4; ++v) {
      int fi = v * 256 + tid;           // 0..1023
      int jr = fi >> 4;                 // 0..63 (j row)
      int c4 = (fi & 15) << 2;          // i offset (coalesced global read)
      float4 vw = *reinterpret_cast<const float4*>(
          &W[((size_t)(e * kJ + j0 + jr)) * kI + i0 + c4]);
      split_f32(vw.x, sBhi[c4 + 0][jr], sBlo[c4 + 0][jr]);
      split_f32(vw.y, sBhi[c4 + 1][jr], sBlo[c4 + 1][jr]);
      split_f32(vw.z, sBhi[c4 + 2][jr], sBlo[c4 + 2][jr]);
      split_f32(vw.w, sBhi[c4 + 3][jr], sBlo[c4 + 3][jr]);
    }

    __syncthreads();

    // ---- Two K=32 steps per chunk; 12 WMMAs per step (2x2 tile x 3 products).
#pragma unroll
    for (int js = 0; js < TJ; js += 32) {
      // A fragments (16x32 f16): lanes 0-15 row M, K {0..7,16..23};
      // lanes 16-31 same rows, K {8..15,24..31}.
      const int ra = wm * 32 + ln;
      const int ja = js + lh * 8;
      v16h Ah0 = frag16(&sAhi[ra][ja],      &sAhi[ra][ja + 16]);
      v16h Al0 = frag16(&sAlo[ra][ja],      &sAlo[ra][ja + 16]);
      v16h Ah1 = frag16(&sAhi[ra + 16][ja], &sAhi[ra + 16][ja + 16]);
      v16h Al1 = frag16(&sAlo[ra + 16][ja], &sAlo[ra + 16][ja + 16]);
      // B fragments (32x16 f16): lane = column N; lanes 0-15 K {0..7,16..23},
      // lanes 16-31 K {8..15,24..31} (i-major sB).
      const int cb = wn * 32 + ln;
      const int jb = js + lh * 8;
      v16h Bh0 = frag16(&sBhi[cb][jb],      &sBhi[cb][jb + 16]);
      v16h Bl0 = frag16(&sBlo[cb][jb],      &sBlo[cb][jb + 16]);
      v16h Bh1 = frag16(&sBhi[cb + 16][jb], &sBhi[cb + 16][jb + 16]);
      v16h Bl1 = frag16(&sBlo[cb + 16][jb], &sBlo[cb + 16][jb + 16]);

#define WMMA_F16(ACC, A, Bm)                                                   \
  ACC = __builtin_amdgcn_wmma_f32_16x16x32_f16(false, (A), false, (Bm),        \
                                               (short)0, (ACC), false, false)
      WMMA_F16(acc00, Ah0, Bh0); WMMA_F16(acc00, Ah0, Bl0); WMMA_F16(acc00, Al0, Bh0);
      WMMA_F16(acc01, Ah0, Bh1); WMMA_F16(acc01, Ah0, Bl1); WMMA_F16(acc01, Al0, Bh1);
      WMMA_F16(acc10, Ah1, Bh0); WMMA_F16(acc10, Ah1, Bl0); WMMA_F16(acc10, Al1, Bh0);
      WMMA_F16(acc11, Ah1, Bh1); WMMA_F16(acc11, Ah1, Bl1); WMMA_F16(acc11, Al1, Bh1);
#undef WMMA_F16
    }

    __syncthreads();
  }

  // ---- Epilogue: scatter-add rows to out[b, Ind[b,e,k], i].
  // C layout: VGPR r, lanes 0-15 = (M=r, N=lane), lanes 16-31 = (M=8+r).
  auto scatter_acc = [&](const v8f& c, int mi, int ni) {
    const int colg  = i0 + wn * 32 + ni * 16 + ln;
    const int rbase = wm * 32 + mi * 16 + lh * 8;
#pragma unroll
    for (int r = 0; r < 8; ++r) {
      const int t = sInd[rbase + r];
      float* dst = out + ((size_t)b * kT + t) * kI + colg;
      __hip_atomic_fetch_add(dst, c[r], __ATOMIC_RELAXED,
                             __HIP_MEMORY_SCOPE_AGENT);
    }
  };
  scatter_acc(acc00, 0, 0);
  scatter_acc(acc01, 0, 1);
  scatter_acc(acc10, 1, 0);
  scatter_acc(acc11, 1, 1);
}

extern "C" void kernel_launch(void* const* d_in, const int* in_sizes, int n_in,
                              void* d_out, int out_size, void* d_ws, size_t ws_size,
                              hipStream_t stream) {
  const float* Y   = (const float*)d_in[0];
  const float* W   = (const float*)d_in[1];
  const int*   Ind = (const int*)d_in[2];
  float* out = (float*)d_out;

  int n4 = out_size / 4;  // kB*kT*kI floats -> float4 count
  zero_out_kernel<<<(n4 + 255) / 256, 256, 0, stream>>>((float4*)out, n4);

  dim3 grid(kI / TN, kK / TM, kB * kE);  // (8, 4, 64)
  gemm_scatter_kernel<<<grid, 256, 0, stream>>>(Y, W, Ind, out);
}
